// RegRNN_40441412059511
// MI455X (gfx1250) — matrix operand order repaired and checked by reading, not compiled, not run on hardware
//
#include <hip/hip_runtime.h>
#include <hip/hip_bf16.h>
#include <stdint.h>

// ---------------------------------------------------------------------------
// RegRNN (L0 hard-concrete gated Elman RNN), MI455X / gfx1250, wave32 + WMMA
// B=64, T=512, I=1024, H=1024  (fp32 in/out; bf16 WMMA w/ f32 accumulate)
// v2.1: weight tiles staged in LDS by the Tensor Data Mover (6-arg builtin),
//       double-buffered, synchronized with s_wait_tensorcnt + barriers.
// ---------------------------------------------------------------------------

#define Bc 64
#define Tc 512
#define Ic 1024
#define Hc 1024

typedef __attribute__((ext_vector_type(16))) __bf16    v16bf;
typedef __attribute__((ext_vector_type(8)))  float     v8f;
typedef __attribute__((ext_vector_type(4)))  float     float4v;
typedef __attribute__((ext_vector_type(4)))  unsigned  uint4v;
typedef __attribute__((ext_vector_type(4)))  unsigned  u32x4;
typedef __attribute__((ext_vector_type(8)))  int       i32x8;
typedef __attribute__((ext_vector_type(4)))  int       i32x4;

union BF16Frag {           // 32 bytes: one wave32 bf16 A or B fragment per lane
    v16bf  v;
    uint4v q[2];
};

__device__ __forceinline__ float hc_gate(float m) {
    // deterministic hard-concrete gate: clip(sigmoid(m)*1.2 - 0.1, 0, 1)
    float s = 1.0f / (1.0f + expf(-m));
    float g = s * 1.2f - 0.1f;
    return fminf(fmaxf(g, 0.0f), 1.0f);
}

// LDS byte offset of a __shared__ object (generic LDS addr: offset in [31:0])
__device__ __forceinline__ uint32_t lds_offset(const void* p) {
    return (uint32_t)(uintptr_t)p;
}

// ---------------------------------------------------------------------------
// TDM: async 2D bf16 tile load Global -> LDS.
//   tile:   rows x 32 elements (k contiguous), row stride = row_stride elems
//   dest:   contiguous rows*64 bytes at lds_off
// D# built per CDNA5 ISA sec 8 (group0: count/lds/global/type, group1: dims).
// ---------------------------------------------------------------------------
__device__ __forceinline__ void tdm_load_tile_bf16(uint32_t lds_off,
                                                   const __bf16* gptr,
                                                   int rows, int row_stride) {
    const uint64_t ga = (uint64_t)(uintptr_t)gptr;
    u32x4 g0;
    g0[0] = 1u;                                           // count=1, no gather
    g0[1] = lds_off;                                      // lds_addr
    g0[2] = (uint32_t)ga;                                 // global_addr[31:0]
    g0[3] = (uint32_t)((ga >> 32) & 0x01FFFFFFu) | (2u << 30); // ga[56:32]|type=2
    i32x8 g1;
    g1[0] = (int)0x00010000;           // wg_mask=0, data_size=1 (2B), no flags
    g1[1] = (int)0xFFFF0000;           // abar=0 | tensor_dim0[15:0]=0xFFFF
    g1[2] = (int)0xFFFF7FFF;           // tensor_dim0[31:16]=0x7FFF | td1[15:0]
    g1[3] = (int)((32u << 16) | 0x7FFFu); // td1[31:16] | tile_dim0 = 32
    g1[4] = (int)(rows & 0xFFFF);      // tile_dim1 = rows, tile_dim2 = 0
    g1[5] = row_stride;                // tensor_dim0_stride[31:0]
    g1[6] = 0;                         // stride hi | tensor_dim1_stride lo
    g1[7] = 0;
    const i32x4 gz4 = {0, 0, 0, 0};               // groups 2/3 unused (2D tile)
    const i32x8 gz8 = {0, 0, 0, 0, 0, 0, 0, 0};   // trailing group unused
    __builtin_amdgcn_tensor_load_to_lds(g0, g1, gz4, gz4, gz8, 0);
}

// ---------------------------------------------------------------------------
// Kernel 1: gate weights -> bf16, fold both biases into one f32 vector
// ---------------------------------------------------------------------------
__global__ void __launch_bounds__(256)
gate_pack_kernel(const float* __restrict__ w_ih, const float* __restrict__ w_ih_mask,
                 const float* __restrict__ w_hh, const float* __restrict__ w_hh_mask,
                 const float* __restrict__ b_ih, const float* __restrict__ b_ih_mask,
                 const float* __restrict__ b_hh, const float* __restrict__ b_hh_mask,
                 __bf16* __restrict__ Wih, __bf16* __restrict__ Whh,
                 float* __restrict__ bias)
{
    int i = blockIdx.x * blockDim.x + threadIdx.x;
    if (i < Hc * Ic) {
        Wih[i] = (__bf16)(hc_gate(w_ih_mask[i]) * w_ih[i]);
        Whh[i] = (__bf16)(hc_gate(w_hh_mask[i]) * w_hh[i]);
    }
    if (i < Hc) {
        bias[i] = hc_gate(b_ih_mask[i]) * b_ih[i] + hc_gate(b_hh_mask[i]) * b_hh[i];
    }
}

// ---------------------------------------------------------------------------
// Kernel 2: zero-init h0 (bf16, B*H elements) -- must run every launch
// ---------------------------------------------------------------------------
__global__ void __launch_bounds__(256)
zero_h_kernel(uint32_t* __restrict__ p, int n_dwords)
{
    int i = blockIdx.x * blockDim.x + threadIdx.x;
    if (i < n_dwords) p[i] = 0u;
}

// ---------------------------------------------------------------------------
// Kernel 3: x_proj = seq @ Wih^T + bias -> d_out [B,T,H] (in-place slot).
// Block: 8 waves = 128(M) x 64(N) macro-tile; wave: 16(M) x 64(N).
// B tile (64 n-rows x 32 k) TDM-staged in LDS, double buffered; all 8 waves
// share it. A = f32 seq, converted to bf16 in-register.
// ---------------------------------------------------------------------------
__global__ void __launch_bounds__(256)
xproj_wmma_kernel(const float* __restrict__ seq, const __bf16* __restrict__ Wih,
                  const float* __restrict__ bias, float* __restrict__ out)
{
    __shared__ __align__(128) __bf16 bsm[2][64 * 32];   // 2 x 4 KB

    const int lane = threadIdx.x & 31;
    const int wave = threadIdx.x >> 5;
    const int m0   = ((blockIdx.x >> 4) << 7) + (wave << 4); // 256 M-blocks
    const int n0   = (blockIdx.x & 15) << 6;                 // 16 N-blocks
    const int l16  = lane & 15;
    const int lh   = lane >> 4;
    const int akb  = lh << 4;                                // A lane K base

    const v8f zero = {0.f,0.f,0.f,0.f,0.f,0.f,0.f,0.f};
    v8f acc[4] = {zero, zero, zero, zero};

    const float* aptr = seq + (size_t)(m0 + l16) * Ic + akb;
    const __bf16* bsrc = Wih + (size_t)n0 * Ic;

    if (wave == 0)
        tdm_load_tile_bf16(lds_offset(&bsm[0][0]), bsrc, 64, Ic);

    for (int i = 0; i < Ic / 32; ++i) {
        const int kk = i << 5;
        if (wave == 0) __builtin_amdgcn_s_wait_tensorcnt(0);
        __syncthreads();                       // tile i visible to all waves
        if (wave == 0 && (i + 1) < Ic / 32)
            tdm_load_tile_bf16(lds_offset(&bsm[(i + 1) & 1][0]),
                               bsrc + kk + 32, 64, Ic);

        // A fragment: row m0+l16; elems 0..15 -> K=akb+kk+0..15 contiguous
        float4v f0 = *(const float4v*)(aptr + kk);
        float4v f1 = *(const float4v*)(aptr + kk + 4);
        float4v f2 = *(const float4v*)(aptr + kk + 8);
        float4v f3 = *(const float4v*)(aptr + kk + 12);
        BF16Frag a;
        a.v[0]  = (__bf16)f0.x;  a.v[1]  = (__bf16)f0.y;
        a.v[2]  = (__bf16)f0.z;  a.v[3]  = (__bf16)f0.w;
        a.v[4]  = (__bf16)f1.x;  a.v[5]  = (__bf16)f1.y;
        a.v[6]  = (__bf16)f1.z;  a.v[7]  = (__bf16)f1.w;
        a.v[8]  = (__bf16)f2.x;  a.v[9]  = (__bf16)f2.y;
        a.v[10] = (__bf16)f2.z;  a.v[11] = (__bf16)f2.w;
        a.v[12] = (__bf16)f3.x;  a.v[13] = (__bf16)f3.y;
        a.v[14] = (__bf16)f3.z;  a.v[15] = (__bf16)f3.w;

        const char* base = (const char*)&bsm[i & 1][0];
        #pragma unroll
        for (int s = 0; s < 4; ++s) {
            // B fragment: col (s*16+l16), K = lh*16 + 0..15 contiguous
            const uint4v* bp =
                (const uint4v*)(base + ((s << 4) + l16) * 64 + (lh << 5));
            BF16Frag b;
            b.q[0] = bp[0];
            b.q[1] = bp[1];
            acc[s] = __builtin_amdgcn_wmma_f32_16x16x32_bf16(
                         false, a.v, false, b.v, (short)0, acc[s], false, false);
        }
        __syncthreads();                 // all reads of tile i done
    }

    #pragma unroll
    for (int s = 0; s < 4; ++s) {
        const int n  = n0 + (s << 4) + l16;
        const float bi = bias[n];
        #pragma unroll
        for (int v = 0; v < 8; ++v) {
            const int m = m0 + v + (lh << 3);          // ISA f32 C/D layout
            out[(size_t)m * Hc + n] = acc[s][v] + bi;
        }
    }
}

// ---------------------------------------------------------------------------
// Kernel 4: one recurrent step: out[:,t,:] = tanh(out[:,t,:] + hprev @ Whh^T)
// Grid: 8 blocks x 8 waves. Block covers 128 N cols; B tile (128 x 32 bf16)
// TDM-staged in LDS, double buffered. Wave: m0=(wave&3)*16, n half=(wave>>2).
// ---------------------------------------------------------------------------
__global__ void __launch_bounds__(256)
rnn_step_wmma_kernel(const __bf16* __restrict__ hprev, __bf16* __restrict__ hnext,
                     const __bf16* __restrict__ Whh, float* __restrict__ out,
                     int t, int is_last)
{
    __shared__ __align__(128) __bf16 bsm[2][128 * 32];  // 2 x 8 KB

    const int lane = threadIdx.x & 31;
    const int wave = threadIdx.x >> 5;
    const int n0b  = blockIdx.x << 7;            // block N base (128 wide)
    const int m0   = (wave & 3) << 4;            // batch tile (0,16,32,48)
    const int nw   = (wave >> 2) << 6;           // 0 or 64 within block tile
    const int l16  = lane & 15;
    const int lh   = lane >> 4;
    const int akb  = lh << 4;

    const v8f zero = {0.f,0.f,0.f,0.f,0.f,0.f,0.f,0.f};
    v8f acc[4] = {zero, zero, zero, zero};

    const __bf16* aptr = hprev + (size_t)(m0 + l16) * Hc + akb;
    const __bf16* bsrc = Whh + (size_t)n0b * Hc;

    if (wave == 0)
        tdm_load_tile_bf16(lds_offset(&bsm[0][0]), bsrc, 128, Hc);

    for (int i = 0; i < Hc / 32; ++i) {
        const int kk = i << 5;
        if (wave == 0) __builtin_amdgcn_s_wait_tensorcnt(0);
        __syncthreads();
        if (wave == 0 && (i + 1) < Hc / 32)
            tdm_load_tile_bf16(lds_offset(&bsm[(i + 1) & 1][0]),
                               bsrc + kk + 32, 128, Hc);

        BF16Frag a;
        a.q[0] = *(const uint4v*)(aptr + kk);        // K = akb+kk .. +7
        a.q[1] = *(const uint4v*)(aptr + kk + 8);    // K = akb+kk+8 .. +15

        const char* base = (const char*)&bsm[i & 1][0];
        #pragma unroll
        for (int s = 0; s < 4; ++s) {
            const uint4v* bp =
                (const uint4v*)(base + (nw + (s << 4) + l16) * 64 + (lh << 5));
            BF16Frag b;
            b.q[0] = bp[0];
            b.q[1] = bp[1];
            acc[s] = __builtin_amdgcn_wmma_f32_16x16x32_bf16(
                         false, a.v, false, b.v, (short)0, acc[s], false, false);
        }
        __syncthreads();
    }

    #pragma unroll
    for (int s = 0; s < 4; ++s) {
        const int n = n0b + nw + (s << 4) + l16;
        #pragma unroll
        for (int v = 0; v < 8; ++v) {
            const int b   = m0 + v + (lh << 3);                 // batch index
            const size_t idx = ((size_t)b * Tc + t) * Hc + n;   // x_proj slot
            const float hv = tanhf(acc[s][v] + out[idx]);
            out[idx] = hv;                                      // output[b,t,n]
            hnext[(size_t)b * Hc + n] = (__bf16)hv;
            if (is_last)                                        // h_last tail
                out[(size_t)Bc * Tc * Hc + (size_t)b * Hc + n] = hv;
        }
    }
}

// ---------------------------------------------------------------------------
extern "C" void kernel_launch(void* const* d_in, const int* in_sizes, int n_in,
                              void* d_out, int out_size, void* d_ws, size_t ws_size,
                              hipStream_t stream) {
    const float* seq       = (const float*)d_in[0];
    const float* w_ih      = (const float*)d_in[1];
    const float* w_ih_mask = (const float*)d_in[2];
    const float* w_hh      = (const float*)d_in[3];
    const float* w_hh_mask = (const float*)d_in[4];
    const float* b_ih      = (const float*)d_in[5];
    const float* b_ih_mask = (const float*)d_in[6];
    const float* b_hh      = (const float*)d_in[7];
    const float* b_hh_mask = (const float*)d_in[8];
    float* out = (float*)d_out;

    // workspace layout (all regions 256B-aligned; total ~4.5 MB)
    char* ws = (char*)d_ws;
    __bf16* Wih  = (__bf16*)(ws);                              // 2 MB
    __bf16* Whh  = (__bf16*)(ws + (size_t)(2u << 20));         // 2 MB
    float*  bias = (float*) (ws + (size_t)(4u << 20));         // 4 KB
    __bf16* hA   = (__bf16*)(ws + (size_t)(4u << 20) + 65536); // 128 KB
    __bf16* hB   = (__bf16*)(ws + (size_t)(4u << 20) + 65536 + Bc * Hc * 2);

    // 1) gate + pack weights, fold biases
    gate_pack_kernel<<<(Hc * Ic + 255) / 256, 256, 0, stream>>>(
        w_ih, w_ih_mask, w_hh, w_hh_mask, b_ih, b_ih_mask, b_hh, b_hh_mask,
        Wih, Whh, bias);

    // 2) h0 = 0  (B*H bf16 = 32768 dwords)
    zero_h_kernel<<<(Bc * Hc / 2 + 255) / 256, 256, 0, stream>>>(
        (uint32_t*)hA, Bc * Hc / 2);

    // 3) x_proj -> d_out[B,T,H]   (32768x1024x1024 bf16 WMMA GEMM)
    xproj_wmma_kernel<<<4096, 256, 0, stream>>>(seq, Wih, bias, out);

    // 4) T sequential recurrent steps, in-place on d_out, h ping-pongs in ws
    for (int t = 0; t < Tc; ++t) {
        const __bf16* hp = (t & 1) ? hB : hA;
        __bf16*       hn = (t & 1) ? hA : hB;
        rnn_step_wmma_kernel<<<8, 256, 0, stream>>>(hp, hn, Whh, out, t,
                                                    (t == Tc - 1) ? 1 : 0);
    }
}
